// SANet_558345748484
// MI455X (gfx1250) — compile-verified, matrix operand order checked by reading, not run on hardware
//
#include <hip/hip_runtime.h>
#include <hip/hip_bf16.h>
#include <cstdint>
#include <cstddef>

// ---------------------------------------------------------------------------
// SANet attention for MI455X (gfx1250), wave32 + WMMA f16 (16x16x32, f32 acc)
// ---------------------------------------------------------------------------

typedef _Float16 half_t;
typedef __attribute__((ext_vector_type(16))) _Float16 v16h;
typedef __attribute__((ext_vector_type(8)))  _Float16 v8h;
typedef __attribute__((ext_vector_type(8)))  float    v8f;

#define C_DIM 512
#define N_DIM 4096
#define B_DIM 4
#define SA_EPS 1e-5f

// Build a 16-bit A/B fragment (16x32 / 32x16, K=32) from a row-major [R][stride]
// f16 buffer. Per ISA 7.12.2: lane L (r=L&15, hh=L>>4) holds row/col r with
// K = k0 + hh*8 + [0,8)  and  k0 + 16 + hh*8 + [0,8). Both chunks are 16B aligned.
__device__ __forceinline__ v16h load_frag(const half_t* __restrict__ base,
                                          int row, int stride, int k0, int hh) {
  const half_t* p = base + (size_t)row * stride + k0 + hh * 8;
  v8h lo = *reinterpret_cast<const v8h*>(p);
  v8h hi = *reinterpret_cast<const v8h*>(p + 16);
  v16h r;
#pragma unroll
  for (int i = 0; i < 8; ++i) { r[i] = lo[i]; r[i + 8] = hi[i]; }
  return r;
}

__device__ __forceinline__ v8f wmma32(v16h a, v16h b, v8f c) {
  // emits v_wmma_f32_16x16x32_f16
  return __builtin_amdgcn_wmma_f32_16x16x32_f16(false, a, false, b,
                                                (short)0, c, false, false);
}

// ---------------------------------------------------------------------------
// Kernel: per-(tensor,b,c) mean and unbiased rstd over the 4096 spatial dim
// ---------------------------------------------------------------------------
__global__ void k_stats(const float* __restrict__ content,
                        const float* __restrict__ style,
                        float* __restrict__ meanBuf,
                        float* __restrict__ rstdBuf) {
  int c = blockIdx.x, b = blockIdx.y, z = blockIdx.z;
  const float* src = (z ? style : content) + ((size_t)b * C_DIM + c) * N_DIM;
  float s = 0.f, q = 0.f;
  for (int i = threadIdx.x; i < N_DIM; i += 256) {
    float v = src[i];
    s += v; q += v * v;
  }
#pragma unroll
  for (int m = 16; m >= 1; m >>= 1) { s += __shfl_xor(s, m); q += __shfl_xor(q, m); }
  __shared__ float ss[8], sq[8];
  int w = threadIdx.x >> 5;
  if ((threadIdx.x & 31) == 0) { ss[w] = s; sq[w] = q; }
  __syncthreads();
  if (threadIdx.x == 0) {
    float S = 0.f, Q = 0.f;
    for (int i = 0; i < 8; ++i) { S += ss[i]; Q += sq[i]; }
    float mean = S / (float)N_DIM;
    float var  = (Q - (float)N_DIM * mean * mean) / (float)(N_DIM - 1);
    size_t o = ((size_t)z * B_DIM + b) * C_DIM + c;
    meanBuf[o] = mean;
    rstdBuf[o] = rsqrtf(var + SA_EPS);
  }
}

// ---------------------------------------------------------------------------
// Kernel: f32 -> f16 weight conversion
// ---------------------------------------------------------------------------
__global__ void k_wcvt(const float* __restrict__ src, half_t* __restrict__ dst, int n) {
  int i = blockIdx.x * 256 + threadIdx.x;
  if (i < n) dst[i] = (half_t)src[i];
}

// ---------------------------------------------------------------------------
// Kernel: (optional MVN) + transpose [b][c][n] f32 -> [b][n][c] f16
// 64x64 LDS tile transpose so downstream WMMA fragments are contiguous.
// ---------------------------------------------------------------------------
__global__ void k_pack(const float* __restrict__ src,
                       const float* __restrict__ meanBuf,   // base for this tensor, or null
                       const float* __restrict__ rstdBuf,
                       half_t* __restrict__ dst) {
  __shared__ float tile[64 * 65];
  int n0 = blockIdx.x * 64, c0 = blockIdx.y * 64, b = blockIdx.z;
#pragma unroll
  for (int i = 0; i < 16; ++i) {
    int idx = threadIdx.x + i * 256;
    int cr = idx >> 6, nc = idx & 63;
    int c = c0 + cr;
    float v = src[((size_t)b * C_DIM + c) * N_DIM + n0 + nc];
    if (meanBuf)
      v = (v - meanBuf[(size_t)b * C_DIM + c]) * rstdBuf[(size_t)b * C_DIM + c];
    tile[cr * 65 + nc] = v;
  }
  __syncthreads();
#pragma unroll
  for (int i = 0; i < 16; ++i) {
    int idx = threadIdx.x + i * 256;
    int nr = idx >> 6, cc = idx & 63;
    dst[((size_t)b * N_DIM + n0 + nr) * C_DIM + c0 + cc] = (half_t)tile[cc * 65 + nr];
  }
}

// ---------------------------------------------------------------------------
// Kernel: conv1x1 as WMMA GEMM.  D[o,n] = sum_c W[o,c] * Xt[n,c]  (+bias)
//   MODE 0: store f16 transposed  [b][n][C]   (A layout for attention: F, G)
//   MODE 1: store f16 natural     [b][C][n]   (B layout for O-GEMM: H)
//   MODE 2: f32 out + bias + residual add     (final conv)
// Block = 8 waves; wave w owns output tile (o0 = (blockIdx.y*8+w)*16, n0).
// ---------------------------------------------------------------------------
template <int MODE>
__global__ void __launch_bounds__(256)
k_gemm(const half_t* __restrict__ W,       // [C][C] f16, row-major (o,c)
       const half_t* __restrict__ Xt,      // [B][N][C] f16
       const float*  __restrict__ bias,    // [C]
       half_t* __restrict__ outH,
       const float* __restrict__ addsrc,   // MODE 2: content [B][C][N]
       float* __restrict__ outF) {         // MODE 2
  int lane = threadIdx.x & 31, r = lane & 15, hh = lane >> 4;
  int w = threadIdx.x >> 5;
  int n0 = blockIdx.x * 16;
  int o0 = (blockIdx.y * 8 + w) * 16;
  int b  = blockIdx.z;
  const half_t* Xb = Xt + (size_t)b * N_DIM * C_DIM;
  v8f acc = {};
#pragma unroll
  for (int kc = 0; kc < 16; ++kc) {                      // K = 512 = 16 x 32
    v16h a  = load_frag(W,  o0 + r, C_DIM, kc * 32, hh); // A: rows of W
    v16h bb = load_frag(Xb, n0 + r, C_DIM, kc * 32, hh); // B: cols = rows of Xt
    acc = wmma32(a, bb, acc);
  }
  // C/D layout: VGPR v -> row o = o0 + v + 8*hh, lane -> col n = n0 + r
  if (MODE == 0) {
    v8h hv;
#pragma unroll
    for (int v = 0; v < 8; ++v) hv[v] = (half_t)(acc[v] + bias[o0 + v + 8 * hh]);
    *reinterpret_cast<v8h*>(outH + ((size_t)b * N_DIM + n0 + r) * C_DIM + o0 + 8 * hh) = hv;
  } else if (MODE == 1) {
#pragma unroll
    for (int v = 0; v < 8; ++v) {
      int o = o0 + v + 8 * hh;
      outH[((size_t)b * C_DIM + o) * N_DIM + n0 + r] = (half_t)(acc[v] + bias[o]);
    }
  } else {
#pragma unroll
    for (int v = 0; v < 8; ++v) {
      int o = o0 + v + 8 * hh;
      size_t idx = ((size_t)b * C_DIM + o) * N_DIM + n0 + r;
      outF[idx] = acc[v] + bias[o] + addsrc[idx];
    }
  }
}

// ---------------------------------------------------------------------------
// Kernel: attention for one (batch, 16-query tile).
//   S[q,s] = sum_c F[q,c] G[s,c]; P = softmax_s(S); O[q,c] = sum_s P[q,s] H[c,s]
// Sweep 1: exact row-max via WMMA (discard S).
// Sweeps 2+3 (x4 segments of 1024 s): recompute S into 64KB LDS, then build
// P fragments (exp in f32, cast f16 -- the LDS round trip is also the
// C/D -> A transpose) and accumulate O via WMMA. Wave w owns channels
// [w*64, w*64+64). Normalize by row sum at the store.
// ---------------------------------------------------------------------------
__global__ void __launch_bounds__(256)
k_attn(const half_t* __restrict__ Ft,   // [B][N][C] f16
       const half_t* __restrict__ Gt,   // [B][N][C] f16
       const half_t* __restrict__ Hc,   // [B][C][N] f16
       half_t* __restrict__ Ot) {       // [B][N][C] f16 out
  __shared__ float Slds[16 * 1024];     // 64 KB; first 128 floats double as wmax
  int lane = threadIdx.x & 31, r = lane & 15, hh = lane >> 4;
  int w = threadIdx.x >> 5;
  int q0 = blockIdx.x * 16;
  int b  = blockIdx.y;
  const half_t* Fb = Ft + (size_t)b * N_DIM * C_DIM;
  const half_t* Gb = Gt + (size_t)b * N_DIM * C_DIM;
  const half_t* Hb = Hc + (size_t)b * C_DIM * N_DIM;

  // Stationary A fragments: 16 query rows x K=512 (16 slices -> 128 VGPRs)
  v16h afrag[16];
#pragma unroll
  for (int kc = 0; kc < 16; ++kc) afrag[kc] = load_frag(Fb, q0 + r, C_DIM, kc * 32, hh);

  // ---- Sweep 1: exact global row max (each wave scans 32 s-tiles) ----
  float rmax[8];
#pragma unroll
  for (int v = 0; v < 8; ++v) rmax[v] = -3.0e38f;
  for (int st = w; st < 256; st += 8) {
    v8f s = {};
#pragma unroll
    for (int kc = 0; kc < 16; ++kc) {
      v16h bf = load_frag(Gb, st * 16 + r, C_DIM, kc * 32, hh);
      s = wmma32(afrag[kc], bf, s);
    }
#pragma unroll
    for (int v = 0; v < 8; ++v) rmax[v] = fmaxf(rmax[v], s[v]);
  }
#pragma unroll
  for (int v = 0; v < 8; ++v) {
#pragma unroll
    for (int m = 1; m < 16; m <<= 1) rmax[v] = fmaxf(rmax[v], __shfl_xor(rmax[v], m));
  }
  if (r == 0) {
#pragma unroll
    for (int v = 0; v < 8; ++v) Slds[w * 16 + v + 8 * hh] = rmax[v];  // wmax[w][row]
  }
  __syncthreads();
  float grmax = Slds[r];                      // row max for row r (A-layout row)
  for (int w2 = 1; w2 < 8; ++w2) grmax = fmaxf(grmax, Slds[w2 * 16 + r]);
  __syncthreads();                            // release wmax region before S writes

  // ---- Sweeps 2+3: 4 segments of 1024 s-columns ----
  v8f oacc[4] = {};
  float lsum = 0.f;
  int c0 = w * 64;
  for (int seg = 0; seg < 4; ++seg) {
    int sbase = seg * 1024;
    for (int j = w; j < 64; j += 8) {         // recompute S tiles -> LDS
      int st = (sbase >> 4) + j;
      v8f s = {};
#pragma unroll
      for (int kc = 0; kc < 16; ++kc) {
        v16h bf = load_frag(Gb, st * 16 + r, C_DIM, kc * 32, hh);
        s = wmma32(afrag[kc], bf, s);
      }
#pragma unroll
      for (int v = 0; v < 8; ++v) Slds[(v + 8 * hh) * 1024 + j * 16 + r] = s[v];
    }
    __syncthreads();
    for (int ch = 0; ch < 32; ++ch) {         // 32-wide P chunks
      int k0 = ch * 32;
      const float* srow = &Slds[r * 1024 + k0 + hh * 8];
      v16h pa;
      float ls = 0.f;
#pragma unroll
      for (int i = 0; i < 8; ++i) { float e = __expf(srow[i]      - grmax); pa[i]     = (half_t)e; ls += e; }
#pragma unroll
      for (int i = 0; i < 8; ++i) { float e = __expf(srow[16 + i] - grmax); pa[8 + i] = (half_t)e; ls += e; }
      lsum += ls;
      int sg = sbase + k0;
#pragma unroll
      for (int ci = 0; ci < 4; ++ci) {        // B from H natural layout [c][s]
        v16h bf = load_frag(Hb, c0 + ci * 16 + r, N_DIM, sg, hh);
        oacc[ci] = wmma32(pa, bf, oacc[ci]);
      }
    }
    __syncthreads();
  }
  lsum += __shfl_xor(lsum, 16);               // fold the two K-halves
  float inv[8];
#pragma unroll
  for (int v = 0; v < 8; ++v) inv[v] = 1.0f / __shfl(lsum, v + 8 * hh);
#pragma unroll
  for (int ci = 0; ci < 4; ++ci) {
#pragma unroll
    for (int v = 0; v < 8; ++v) {
      Ot[((size_t)b * N_DIM + q0 + v + 8 * hh) * C_DIM + c0 + ci * 16 + r] =
          (half_t)(oacc[ci][v] * inv[v]);
    }
  }
}

// ---------------------------------------------------------------------------
// Host launcher. Workspace layout (~115 MB):
//   mean/rstd (32KB) | 4x f16 weights (2MB) | cn,sn,sr,Ft,Gt,Hc,Ot (7 x 16MB)
// ---------------------------------------------------------------------------
extern "C" void kernel_launch(void* const* d_in, const int* in_sizes, int n_in,
                              void* d_out, int out_size, void* d_ws, size_t ws_size,
                              hipStream_t stream) {
  (void)in_sizes; (void)n_in; (void)out_size; (void)ws_size;
  const float* content = (const float*)d_in[0];
  const float* style   = (const float*)d_in[1];
  const float* f_w = (const float*)d_in[2]; const float* f_b = (const float*)d_in[3];
  const float* g_w = (const float*)d_in[4]; const float* g_b = (const float*)d_in[5];
  const float* h_w = (const float*)d_in[6]; const float* h_b = (const float*)d_in[7];
  const float* o_w = (const float*)d_in[8]; const float* o_b = (const float*)d_in[9];
  float* out = (float*)d_out;

  char* ws = (char*)d_ws;
  size_t off = 0;
  const size_t statsN = (size_t)2 * B_DIM * C_DIM;
  const size_t wN     = (size_t)C_DIM * C_DIM;
  const size_t actN   = (size_t)B_DIM * N_DIM * C_DIM;
  float*  meanBuf = (float*)(ws + off); off += statsN * sizeof(float);
  float*  rstdBuf = (float*)(ws + off); off += statsN * sizeof(float);
  half_t* Whf = (half_t*)(ws + off); off += wN * sizeof(half_t);
  half_t* Whg = (half_t*)(ws + off); off += wN * sizeof(half_t);
  half_t* Whh = (half_t*)(ws + off); off += wN * sizeof(half_t);
  half_t* Who = (half_t*)(ws + off); off += wN * sizeof(half_t);
  half_t* cn  = (half_t*)(ws + off); off += actN * sizeof(half_t);
  half_t* sn  = (half_t*)(ws + off); off += actN * sizeof(half_t);
  half_t* sr  = (half_t*)(ws + off); off += actN * sizeof(half_t);
  half_t* Ftb = (half_t*)(ws + off); off += actN * sizeof(half_t);
  half_t* Gtb = (half_t*)(ws + off); off += actN * sizeof(half_t);
  half_t* Hcb = (half_t*)(ws + off); off += actN * sizeof(half_t);
  half_t* Otb = (half_t*)(ws + off); off += actN * sizeof(half_t);

  // 1) instance-norm statistics
  k_stats<<<dim3(C_DIM, B_DIM, 2), 256, 0, stream>>>(content, style, meanBuf, rstdBuf);

  // 2) f16 weights
  int wn = (int)wN;
  k_wcvt<<<(wn + 255) / 256, 256, 0, stream>>>(f_w, Whf, wn);
  k_wcvt<<<(wn + 255) / 256, 256, 0, stream>>>(g_w, Whg, wn);
  k_wcvt<<<(wn + 255) / 256, 256, 0, stream>>>(h_w, Whh, wn);
  k_wcvt<<<(wn + 255) / 256, 256, 0, stream>>>(o_w, Who, wn);

  // 3) normalize + transpose + f16 pack
  dim3 pg(N_DIM / 64, C_DIM / 64, B_DIM);
  k_pack<<<pg, 256, 0, stream>>>(content, meanBuf, rstdBuf, cn);
  k_pack<<<pg, 256, 0, stream>>>(style, meanBuf + (size_t)B_DIM * C_DIM,
                                 rstdBuf + (size_t)B_DIM * C_DIM, sn);
  k_pack<<<pg, 256, 0, stream>>>(style, nullptr, nullptr, sr);

  // 4) f/g/h conv1x1 via WMMA
  dim3 gg(N_DIM / 16, C_DIM / 128, B_DIM);
  k_gemm<0><<<gg, 256, 0, stream>>>(Whf, cn, f_b, Ftb, nullptr, nullptr);
  k_gemm<0><<<gg, 256, 0, stream>>>(Whg, sn, g_b, Gtb, nullptr, nullptr);
  k_gemm<1><<<gg, 256, 0, stream>>>(Whh, sr, h_b, Hcb, nullptr, nullptr);

  // 5) attention (softmax fully in-LDS, never materialize S in HBM)
  k_attn<<<dim3(N_DIM / 16, B_DIM), 256, 0, stream>>>(Ftb, Gtb, Hcb, Otb);

  // 6) output conv1x1 + bias + residual
  k_gemm<2><<<gg, 256, 0, stream>>>(Who, Otb, o_b, nullptr, content, out);
}
